// HybridVelocityGAT_67053029425712
// MI455X (gfx1250) — matrix operand order repaired and checked by reading, not compiled, not run on hardware
//
#include <hip/hip_runtime.h>
#include <hip/hip_bf16.h>
#include <math.h>

// ---------------------------------------------------------------------------
// HybridVelocityGAT for MI455X (gfx1250, wave32).
// Dense transforms: v_wmma_f32_16x16x32_f16 with f16 operands pre-packed /
// zero-padded so the K-loop is pure global_load_b128 + v_wmma (no guards).
// Edge phase: wave-per-edge gather + wave32 shfl reduction + L2 float atomics.
// ---------------------------------------------------------------------------

typedef __attribute__((ext_vector_type(16))) _Float16 v16h;
typedef __attribute__((ext_vector_type(8)))  _Float16 v8h;
typedef __attribute__((ext_vector_type(8)))  float    v8f;

#define EPSBN 1e-5f
#define XK 96    // padded K for layer-1 input (66 -> 96, multiple of 32)

__device__ __forceinline__ float eluf(float x)   { return x > 0.f ? x : (expf(x) - 1.f); }
__device__ __forceinline__ float leakyf(float x) { return x > 0.f ? x : 0.2f * x; }

// Monotonic float <-> uint encoding for atomicMax-based segment max.
__device__ __forceinline__ unsigned fenc(float f) {
    unsigned u = __float_as_uint(f);
    return (u & 0x80000000u) ? ~u : (u | 0x80000000u);
}
__device__ __forceinline__ float fdec(unsigned u) {
    return (u & 0x80000000u) ? __uint_as_float(u & 0x7FFFFFFFu) : __uint_as_float(~u);
}

// ---------------------------------------------------------------------------
// Weight pre-pack: W[Cout,Cin] f32 -> Wh[CoutPad,Kpad] f16, zero-padded.
// ---------------------------------------------------------------------------
__global__ void pack_weight_f16_kernel(const float* __restrict__ W,
                                       int Cout, int Cin,
                                       _Float16* __restrict__ Wh,
                                       int Kpad, int total) {
    const int tid = blockIdx.x * blockDim.x + threadIdx.x;
    if (tid >= total) return;
    const int r = tid / Kpad, k = tid - r * Kpad;
    Wh[tid] = (r < Cout && k < Cin) ? (_Float16)W[r * Cin + k] : (_Float16)0.f;
}

// ---------------------------------------------------------------------------
// Kernel 1: bio encoder (Linear 64->16, BN, ELU, Linear 16->16, ELU) fused
// with concat into x16[Npad, XK] f16 (0..49 = pca, 50..65 = h_bio, 66.. = 0).
// 256 threads = 16 nodes * 16 output channels, LDS staged.
// ---------------------------------------------------------------------------
__global__ void bio_concat_kernel(const float* __restrict__ x_pca,
                                  const float* __restrict__ x_bio,
                                  const float* __restrict__ W1, const float* __restrict__ b1,
                                  const float* __restrict__ g0, const float* __restrict__ be0,
                                  const float* __restrict__ m0, const float* __restrict__ v0,
                                  const float* __restrict__ W2, const float* __restrict__ b2,
                                  _Float16* __restrict__ x16, int N) {
    __shared__ float sW1[16 * 64];
    __shared__ float sXB[16 * 64];
    __shared__ float sH1[16 * 16];
    const int t   = threadIdx.x;
    const int g0n = blockIdx.x * 16;

    for (int i = t; i < 16 * 64; i += 256) sW1[i] = W1[i];
    for (int i = t; i < 16 * 64; i += 256) {
        int r = i >> 6, c = i & 63;
        int gn = g0n + r;
        sXB[i] = (gn < N) ? x_bio[gn * 64 + c] : 0.f;
    }
    __syncthreads();

    const int node = t >> 4;
    const int j    = t & 15;
    const int gn   = g0n + node;

    float acc = b1[j];
    #pragma unroll 8
    for (int k = 0; k < 64; ++k) acc += sXB[node * 64 + k] * sW1[j * 64 + k];
    acc = (acc - m0[j]) * rsqrtf(v0[j] + EPSBN) * g0[j] + be0[j];
    acc = eluf(acc);
    sH1[node * 16 + j] = acc;
    __syncthreads();

    float acc2 = b2[j];
    #pragma unroll
    for (int k = 0; k < 16; ++k) acc2 += sH1[node * 16 + k] * W2[j * 16 + k];
    acc2 = eluf(acc2);
    if (gn < N) x16[(size_t)gn * XK + 50 + j] = (_Float16)acc2;

    // PCA features (cols 0..49) and zero pad (cols 66..95)
    for (int i = t; i < 16 * 50; i += 256) {
        int r = i / 50, c = i % 50;
        int g = g0n + r;
        if (g < N) x16[(size_t)g * XK + c] = (_Float16)x_pca[g * 50 + c];
    }
    for (int i = t; i < 16 * 30; i += 256) {
        int r = i / 30, c = 66 + i % 30;
        int g = g0n + r;
        if (g < N) x16[(size_t)g * XK + c] = (_Float16)0.f;
    }
}

// ---------------------------------------------------------------------------
// Kernel 2: WMMA linear  Y[nrows,Cout] = X16[.,Kpad] @ Wh^T + bias.
// One wave per 16x16 tile. Operands are pre-padded: inner loop is pure
// b128 loads + v_wmma_f32_16x16x32_f16. Guards only at the final store.
// A fragment per lane: K runs [k0+hi*8, +7] and [k0+16+hi*8, +7] (ISA 7.12.2).
// B fragment per lane: K runs [k0+hi*8, +7] and [k0+16+hi*8, +7] (same as A).
// ---------------------------------------------------------------------------
__global__ void wmma_linear_kernel(const _Float16* __restrict__ X, int ldx,
                                   const _Float16* __restrict__ Wh, int ldw,
                                   const float* __restrict__ bias,
                                   float* __restrict__ Y, int ldy,
                                   int nrows, int mtiles, int Kpad, int Cout) {
    const int wave = threadIdx.x >> 5;
    const int lane = threadIdx.x & 31;
    const int mt   = blockIdx.x * (blockDim.x >> 5) + wave;
    if (mt >= mtiles) return;                 // wave-uniform exit (EXEC stays full)
    const int m0 = mt * 16;
    const int n0 = blockIdx.y * 16;

    const int hi = lane >> 4;
    const int lo = lane & 15;

    const _Float16* __restrict__ xrow = X  + (size_t)(m0 + lo) * ldx + hi * 8;
    const _Float16* __restrict__ wrow = Wh + (size_t)(n0 + lo) * ldw + hi * 8;

    v8f c = {};
    for (int k0 = 0; k0 < Kpad; k0 += 32) {
        v8h a0 = *(const v8h*)(xrow + k0);        // K = k0+hi*8   .. +7
        v8h a1 = *(const v8h*)(xrow + k0 + 16);   // K = k0+16+hi*8 .. +7
        v8h b0 = *(const v8h*)(wrow + k0);        // K = k0+hi*8   .. +7
        v8h b1 = *(const v8h*)(wrow + k0 + 16);   // K = k0+16+hi*8 .. +7
        v16h a = __builtin_shufflevector(a0, a1, 0,1,2,3,4,5,6,7,8,9,10,11,12,13,14,15);
        v16h b = __builtin_shufflevector(b0, b1, 0,1,2,3,4,5,6,7,8,9,10,11,12,13,14,15);
        c = __builtin_amdgcn_wmma_f32_16x16x32_f16(false, a, false, b,
                                                   (short)0, c, false, false);
    }

    const int n = n0 + lo;
    if (n < Cout) {
        float bv = bias[n];
        #pragma unroll
        for (int r = 0; r < 8; ++r) {
            int m = m0 + r + hi * 8;
            if (m < nrows) Y[(size_t)m * ldy + n] = c[r] + bv;
        }
    }
}

// ---------------------------------------------------------------------------
// Kernel 3: wave-per-edge attention logits + segment max (encoded atomicMax).
// ---------------------------------------------------------------------------
__global__ void edge_logits_kernel(const float* __restrict__ xl,
                                   const float* __restrict__ xr,
                                   const int* __restrict__ ei,
                                   int E, int Etot,
                                   const float* __restrict__ att,
                                   float* __restrict__ logits,
                                   unsigned* __restrict__ maxbuf, int H) {
    const int gw   = (int)((blockIdx.x * (unsigned)blockDim.x + threadIdx.x) >> 5);
    const int lane = threadIdx.x & 31;
    if (gw >= Etot) return;                   // wave-uniform

    const int s = (gw < E) ? ei[gw]     : (gw - E);
    const int d = (gw < E) ? ei[E + gw] : (gw - E);
    const int ld = H * 32;

    for (int h = 0; h < H; ++h) {
        float v = xl[(size_t)s * ld + h * 32 + lane] + xr[(size_t)d * ld + h * 32 + lane];
        float p = leakyf(v) * att[h * 32 + lane];
        #pragma unroll
        for (int off = 16; off > 0; off >>= 1) p += __shfl_xor(p, off, 32);
        if (lane == 0) {
            logits[(size_t)gw * H + h] = p;
            atomicMax(&maxbuf[d * H + h], fenc(p));
        }
    }
}

// ---------------------------------------------------------------------------
// Kernel 4: alpha = exp(logit - max[dst]); denom[dst] += alpha (in-place).
// ---------------------------------------------------------------------------
__global__ void alpha_den_kernel(float* __restrict__ logits,
                                 const unsigned* __restrict__ maxbuf,
                                 float* __restrict__ den,
                                 const int* __restrict__ ei,
                                 int E, int Etot, int H) {
    const int tid = blockIdx.x * blockDim.x + threadIdx.x;
    if (tid >= Etot * H) return;
    const int e = tid / H, h = tid - e * H;
    const int d = (e < E) ? ei[E + e] : (e - E);
    float a = expf(logits[tid] - fdec(maxbuf[d * H + h]));
    logits[tid] = a;
    atomicAdd(&den[d * H + h], a);
}

// ---------------------------------------------------------------------------
// Kernel 5: wave-per-edge weighted scatter-add aggregation.
// ---------------------------------------------------------------------------
__global__ void aggregate_kernel(const float* __restrict__ xl,
                                 const float* __restrict__ alpha,
                                 const float* __restrict__ den,
                                 const int* __restrict__ ei,
                                 int E, int Etot,
                                 float* __restrict__ out, int H) {
    const int gw   = (int)((blockIdx.x * (unsigned)blockDim.x + threadIdx.x) >> 5);
    const int lane = threadIdx.x & 31;
    if (gw >= Etot) return;

    const int s = (gw < E) ? ei[gw]     : (gw - E);
    const int d = (gw < E) ? ei[E + gw] : (gw - E);
    const int ld = H * 32;

    for (int h = 0; h < H; ++h) {
        float w = alpha[(size_t)gw * H + h] / (den[d * H + h] + 1e-16f);
        atomicAdd(&out[(size_t)d * ld + h * 32 + lane],
                  xl[(size_t)s * ld + h * 32 + lane] * w);
    }
}

// ---------------------------------------------------------------------------
// Kernel 6: h16 = f16(ELU(BN(agg + bias)))  (f32 in, f16 out for next WMMA).
// ---------------------------------------------------------------------------
__global__ void post_bn_elu_kernel(const float* __restrict__ hin,
                                   _Float16* __restrict__ hout,
                                   const float* __restrict__ bias,
                                   const float* __restrict__ g,
                                   const float* __restrict__ b,
                                   const float* __restrict__ m,
                                   const float* __restrict__ v,
                                   int total, int C) {
    const int tid = blockIdx.x * blockDim.x + threadIdx.x;
    if (tid >= total) return;
    const int c = tid % C;
    float val = hin[tid] + bias[c];
    val = (val - m[c]) * rsqrtf(v[c] + EPSBN) * g[c] + b[c];
    hout[tid] = (_Float16)eluf(val);
}

// ---------------------------------------------------------------------------
// Host side
// ---------------------------------------------------------------------------
extern "C" void kernel_launch(void* const* d_in, const int* in_sizes, int n_in,
                              void* d_out, int out_size, void* d_ws, size_t ws_size,
                              hipStream_t stream) {
    (void)n_in; (void)out_size; (void)ws_size;
    const float* x_pca = (const float*)d_in[0];
    const float* x_bio = (const float*)d_in[1];
    const int*   ei    = (const int*)  d_in[2];
    const float* bioW1 = (const float*)d_in[3];
    const float* biob1 = (const float*)d_in[4];
    const float* bn0_g = (const float*)d_in[5];
    const float* bn0_b = (const float*)d_in[6];
    const float* bn0_m = (const float*)d_in[7];
    const float* bn0_v = (const float*)d_in[8];
    const float* bioW2 = (const float*)d_in[9];
    const float* biob2 = (const float*)d_in[10];
    const float* Wl1   = (const float*)d_in[11];
    const float* bl1   = (const float*)d_in[12];
    const float* Wr1   = (const float*)d_in[13];
    const float* br1   = (const float*)d_in[14];
    const float* att1  = (const float*)d_in[15];
    const float* bias1 = (const float*)d_in[16];
    const float* bn1_g = (const float*)d_in[17];
    const float* bn1_b = (const float*)d_in[18];
    const float* bn1_m = (const float*)d_in[19];
    const float* bn1_v = (const float*)d_in[20];
    const float* Wl2   = (const float*)d_in[21];
    const float* bl2   = (const float*)d_in[22];
    const float* Wr2   = (const float*)d_in[23];
    const float* br2   = (const float*)d_in[24];
    const float* att2  = (const float*)d_in[25];
    const float* bias2 = (const float*)d_in[26];
    const float* bn2_g = (const float*)d_in[27];
    const float* bn2_b = (const float*)d_in[28];
    const float* bn2_m = (const float*)d_in[29];
    const float* bn2_v = (const float*)d_in[30];
    const float* predW = (const float*)d_in[31];
    const float* predb = (const float*)d_in[32];

    const int N    = in_sizes[0] / 50;
    const int E    = in_sizes[2] / 2;
    const int Etot = E + N;
    const int Npad = ((N + 15) / 16) * 16;

    // ---- workspace layout (float units, all offsets multiples of 4)
    float* ws = (float*)d_ws;
    size_t o = 0;
    _Float16* x16    = (_Float16*)(ws + o); o += (size_t)Npad * XK / 2;   // [Npad,96] f16
    _Float16* wh_l1  = (_Float16*)(ws + o); o += 128 * 96 / 2;
    _Float16* wh_r1  = (_Float16*)(ws + o); o += 128 * 96 / 2;
    _Float16* wh_l2  = (_Float16*)(ws + o); o += 32 * 128 / 2;
    _Float16* wh_r2  = (_Float16*)(ws + o); o += 32 * 128 / 2;
    _Float16* wh_pr  = (_Float16*)(ws + o); o += 64 * 32 / 2;
    float* buf_xl1   = ws + o; o += (size_t)Npad * 128;
    float* buf_xr1   = ws + o; o += (size_t)Npad * 128;
    float* buf_alp1  = ws + o; o += (size_t)Etot * 4;        // logits -> alpha
    float* buf_max1  = ws + o; o += (size_t)N * 4;           // encoded u32 max
    float* buf_den1  = ws + o; o += (size_t)N * 4;
    float* buf_h1agg = ws + o; o += (size_t)N * 128;
    _Float16* h1h    = (_Float16*)(ws + o); o += (size_t)Npad * 128 / 2;  // f16 copy
    // ---- layer-2 aliases (source regions dead by the time they're used)
    float*    buf_xl2  = (float*)x16;                         // Npad*32 <= Npad*48
    float*    buf_xr2  = buf_h1agg;                           // free after h1h written
    float*    buf_h2agg = buf_h1agg + (size_t)Npad * 32;      // disjoint from xr2
    _Float16* h2h      = (_Float16*)buf_xl1;                  // free after layer 1
    float*    buf_alp2 = buf_alp1;
    float*    buf_max2 = buf_max1;
    float*    buf_den2 = buf_den1;

    const int mtiles = Npad / 16;
    const dim3 wblk(128);                                     // 4 waves / block
    const int wgx = (mtiles + 3) / 4;
    const int edgeBlocks = (Etot + 7) / 8;                    // 8 waves / block

    // ===================== weight packing (tiny) =====================
    pack_weight_f16_kernel<<<(128 * 96 + 255) / 256, 256, 0, stream>>>(Wl1, 128, 66, wh_l1, 96, 128 * 96);
    pack_weight_f16_kernel<<<(128 * 96 + 255) / 256, 256, 0, stream>>>(Wr1, 128, 66, wh_r1, 96, 128 * 96);
    pack_weight_f16_kernel<<<(32 * 128 + 255) / 256, 256, 0, stream>>>(Wl2, 32, 128, wh_l2, 128, 32 * 128);
    pack_weight_f16_kernel<<<(32 * 128 + 255) / 256, 256, 0, stream>>>(Wr2, 32, 128, wh_r2, 128, 32 * 128);
    pack_weight_f16_kernel<<<(64 * 32 + 255) / 256, 256, 0, stream>>>(predW, 50, 32, wh_pr, 32, 64 * 32);

    // ===================== bio encoder + concat =====================
    bio_concat_kernel<<<(N + 15) / 16, 256, 0, stream>>>(
        x_pca, x_bio, bioW1, biob1, bn0_g, bn0_b, bn0_m, bn0_v,
        bioW2, biob2, x16, N);

    // ===================== GATv2 layer 1 (H=4, C=32) =====================
    hipMemsetAsync(buf_max1, 0, (size_t)N * 4 * sizeof(unsigned), stream);
    hipMemsetAsync(buf_den1, 0, (size_t)N * 4 * sizeof(float), stream);
    hipMemsetAsync(buf_h1agg, 0, (size_t)N * 128 * sizeof(float), stream);

    wmma_linear_kernel<<<dim3(wgx, 8), wblk, 0, stream>>>(
        x16, XK, wh_l1, 96, bl1, buf_xl1, 128, N, mtiles, 96, 128);
    wmma_linear_kernel<<<dim3(wgx, 8), wblk, 0, stream>>>(
        x16, XK, wh_r1, 96, br1, buf_xr1, 128, N, mtiles, 96, 128);

    edge_logits_kernel<<<edgeBlocks, 256, 0, stream>>>(
        buf_xl1, buf_xr1, ei, E, Etot, att1, buf_alp1, (unsigned*)buf_max1, 4);
    alpha_den_kernel<<<((size_t)Etot * 4 + 255) / 256, 256, 0, stream>>>(
        buf_alp1, (const unsigned*)buf_max1, buf_den1, ei, E, Etot, 4);
    aggregate_kernel<<<edgeBlocks, 256, 0, stream>>>(
        buf_xl1, buf_alp1, buf_den1, ei, E, Etot, buf_h1agg, 4);
    post_bn_elu_kernel<<<((size_t)N * 128 + 255) / 256, 256, 0, stream>>>(
        buf_h1agg, h1h, bias1, bn1_g, bn1_b, bn1_m, bn1_v, N * 128, 128);

    // ===================== GATv2 layer 2 (H=1, C=32) =====================
    wmma_linear_kernel<<<dim3(wgx, 2), wblk, 0, stream>>>(
        h1h, 128, wh_l2, 128, bl2, buf_xl2, 32, N, mtiles, 128, 32);
    wmma_linear_kernel<<<dim3(wgx, 2), wblk, 0, stream>>>(
        h1h, 128, wh_r2, 128, br2, buf_xr2, 32, N, mtiles, 128, 32);

    hipMemsetAsync(buf_max2, 0, (size_t)N * sizeof(unsigned), stream);
    hipMemsetAsync(buf_den2, 0, (size_t)N * sizeof(float), stream);
    hipMemsetAsync(buf_h2agg, 0, (size_t)N * 32 * sizeof(float), stream);

    edge_logits_kernel<<<edgeBlocks, 256, 0, stream>>>(
        buf_xl2, buf_xr2, ei, E, Etot, att2, buf_alp2, (unsigned*)buf_max2, 1);
    alpha_den_kernel<<<((size_t)Etot + 255) / 256, 256, 0, stream>>>(
        buf_alp2, (const unsigned*)buf_max2, buf_den2, ei, E, Etot, 1);
    aggregate_kernel<<<edgeBlocks, 256, 0, stream>>>(
        buf_xl2, buf_alp2, buf_den2, ei, E, Etot, buf_h2agg, 1);
    post_bn_elu_kernel<<<((size_t)N * 32 + 255) / 256, 256, 0, stream>>>(
        buf_h2agg, h2h, bias2, bn2_g, bn2_b, bn2_m, bn2_v, N * 32, 32);

    // ===================== prediction head (32 -> 50) =====================
    wmma_linear_kernel<<<dim3(wgx, 4), wblk, 0, stream>>>(
        h2h, 32, wh_pr, 32, predb, (float*)d_out, 50, N, mtiles, 32, 50);
}
